// Attention_66236985639651
// MI455X (gfx1250) — matrix operand order repaired and checked by reading, not compiled, not run on hardware
//
#include <hip/hip_runtime.h>
#include <hip/hip_bf16.h>

typedef __attribute__((ext_vector_type(16))) __bf16 v16bf;
typedef __attribute__((ext_vector_type(8)))  __bf16 v8bf;
typedef __attribute__((ext_vector_type(8)))  float  v8f;

#define DIMX     1024
#define DH       512
#define DIMO     1024
#define NSEQ     8192
#define MASK_VAL (-1e10f)

union V16U { v16bf v; v8bf h[2]; };

// A-fragment (16x32 bf16, ISA 7.12.2): lane m=lane&15, kb=(lane>>4)*8;
// a[0..7] = A[m][kb..kb+7], a[8..15] = A[m][16+kb..16+kb+7]
__device__ inline v16bf load_a_frag(const __bf16* __restrict__ src, int row0, int ld,
                                    int k0, int lane) {
  int m  = lane & 15;
  int kb = (lane >> 4) * 8;
  const __bf16* p = src + (long)(row0 + m) * ld + k0;
  V16U u;
  u.h[0] = *(const v8bf*)(p + kb);
  u.h[1] = *(const v8bf*)(p + 16 + kb);
  return u.v;
}

// B-fragment (32x16 bf16): lane n=lane&15, kb2=(lane>>4)*8;
// b[0..7] = srcT[col0+n][k0+kb2..+7], b[8..15] = srcT[col0+n][k0+16+kb2..+7]  (srcT row-major, stride ldT)
__device__ inline v16bf load_b_frag(const __bf16* __restrict__ srcT, int col0, int ldT,
                                    int k0, int lane) {
  int n   = lane & 15;
  int kb2 = (lane >> 4) * 8;
  const __bf16* p = srcT + (long)(col0 + n) * ldT + k0 + kb2;
  V16U u;
  u.h[0] = *(const v8bf*)(p);
  u.h[1] = *(const v8bf*)(p + 16);
  return u.v;
}

__device__ inline v8f wmma_bf16(v16bf a, v16bf b, v8f c) {
  return __builtin_amdgcn_wmma_f32_16x16x32_bf16(false, a, false, b, (short)0, c,
                                                 false, false);
}

// ---------------- precision conversion / weight transpose ----------------
// 8 elements per thread: 2x b128 load, 1x b128 store (pure bandwidth).
__global__ void cvt_bf16_x8(const float* __restrict__ src, __bf16* __restrict__ dst,
                            int n) {
  int i = (blockIdx.x * blockDim.x + threadIdx.x) * 8;
  if (i < n) {
    const float4* s = (const float4*)(src + i);
    float4 f0 = s[0], f1 = s[1];
    v8bf o;
    o[0] = (__bf16)f0.x; o[1] = (__bf16)f0.y; o[2] = (__bf16)f0.z; o[3] = (__bf16)f0.w;
    o[4] = (__bf16)f1.x; o[5] = (__bf16)f1.y; o[6] = (__bf16)f1.z; o[7] = (__bf16)f1.w;
    *(v8bf*)(dst + i) = o;
  }
}

__global__ void cvt_transpose_bf16(const float* __restrict__ src, __bf16* __restrict__ dst,
                                   int rows, int cols) {
  int i = blockIdx.x * blockDim.x + threadIdx.x;
  if (i < rows * cols) {
    int r = i / cols, c = i % cols;
    dst[(long)c * rows + r] = (__bf16)src[i];
  }
}

// ---------------- QKV projection: qkv = x @ w_qkv + b_qkv ----------------
// Per wave: 32 rows x 64 cols, double-buffered fragment loads.
// q,k stored row-major [NSEQ][DH] bf16; v stored TRANSPOSED [DH][NSEQ] bf16.
__global__ void __launch_bounds__(128)
qkv_proj(const __bf16* __restrict__ xb, const __bf16* __restrict__ wt,
         const float* __restrict__ bias, __bf16* __restrict__ qb,
         __bf16* __restrict__ kbuf, __bf16* __restrict__ vt) {
  const int lane = threadIdx.x & 31;
  const int w    = threadIdx.x >> 5;
  const int rb   = blockIdx.x * 128 + w * 32;
  const int cb   = blockIdx.y * 64;
  const int n16  = lane & 15, hi = lane >> 4;

  v8f acc[2][4];
  #pragma unroll
  for (int mt = 0; mt < 2; ++mt)
    #pragma unroll
    for (int nt = 0; nt < 4; ++nt) acc[mt][nt] = v8f{0,0,0,0,0,0,0,0};

  v16bf a0[2], b0[4], a1[2], b1[4];
  #pragma unroll
  for (int mt = 0; mt < 2; ++mt) a0[mt] = load_a_frag(xb, rb + mt * 16, DIMX, 0, lane);
  #pragma unroll
  for (int nt = 0; nt < 4; ++nt) b0[nt] = load_b_frag(wt, cb + nt * 16, DIMX, 0, lane);

  for (int kc = 0; kc < DIMX; kc += 64) {
    // stage 1: issue loads for kc+32, compute on buffer 0 (kc)
    #pragma unroll
    for (int mt = 0; mt < 2; ++mt)
      a1[mt] = load_a_frag(xb, rb + mt * 16, DIMX, kc + 32, lane);
    #pragma unroll
    for (int nt = 0; nt < 4; ++nt)
      b1[nt] = load_b_frag(wt, cb + nt * 16, DIMX, kc + 32, lane);
    #pragma unroll
    for (int mt = 0; mt < 2; ++mt)
      #pragma unroll
      for (int nt = 0; nt < 4; ++nt)
        acc[mt][nt] = wmma_bf16(a0[mt], b0[nt], acc[mt][nt]);
    // stage 2: issue loads for kc+64 (if any), compute on buffer 1 (kc+32)
    if (kc + 64 < DIMX) {
      #pragma unroll
      for (int mt = 0; mt < 2; ++mt)
        a0[mt] = load_a_frag(xb, rb + mt * 16, DIMX, kc + 64, lane);
      #pragma unroll
      for (int nt = 0; nt < 4; ++nt)
        b0[nt] = load_b_frag(wt, cb + nt * 16, DIMX, kc + 64, lane);
    }
    #pragma unroll
    for (int mt = 0; mt < 2; ++mt)
      #pragma unroll
      for (int nt = 0; nt < 4; ++nt)
        acc[mt][nt] = wmma_bf16(a1[mt], b1[nt], acc[mt][nt]);
  }

  #pragma unroll
  for (int mt = 0; mt < 2; ++mt) {
    #pragma unroll
    for (int nt = 0; nt < 4; ++nt) {
      int col  = cb + nt * 16 + n16;
      float bv = bias[col];
      #pragma unroll
      for (int r = 0; r < 8; ++r) {
        int row = rb + mt * 16 + r + 8 * hi;   // C/D layout: c[r] = C[r+8*hi][n16]
        float v = acc[mt][nt][r] + bv;
        if (col < DH)          qb[(long)row * DH + col]              = (__bf16)v;
        else if (col < 2 * DH) kbuf[(long)row * DH + (col - DH)]     = (__bf16)v;
        else                   vt[(long)(col - 2 * DH) * NSEQ + row] = (__bf16)v;
      }
    }
  }
}

// ---------------- Flash attention (causal, online softmax) ----------------
// Block: 128 threads = 4 waves; 16 query rows per block; each wave owns a
// 128-wide dh slice of the O accumulator. Partial S reduced via LDS.
// K fragments batch-loaded before the S WMMAs; V fragments issued early so
// their latency hides under the LDS reduction + exp softmax work.
__global__ void __launch_bounds__(128)
attn_kernel(const __bf16* __restrict__ qb, const __bf16* __restrict__ kb,
            const __bf16* __restrict__ vt, __bf16* __restrict__ ob) {
  __shared__ float sred[2 * 8 * 4 * 32];                     // [tile][r][wave][lane]
  __shared__ __attribute__((aligned(16))) __bf16 pbuf[16 * 32];

  const int lane  = threadIdx.x & 31;
  const int w     = threadIdx.x >> 5;
  const int qbase = blockIdx.x * 16;
  const int ds0   = w * 128;                                 // this wave's dh slice
  const int n16   = lane & 15, hi = lane >> 4;
  const float scale = 0.044194173824159216f;                 // 1/sqrt(512)

  // Q fragments resident in VGPRs for all kv steps
  v16bf qf[4];
  #pragma unroll
  for (int c = 0; c < 4; ++c)
    qf[c] = load_a_frag(qb, qbase, DH, ds0 + c * 32, lane);

  v8f oa[8];
  float mrow[8], lrow[8];
  #pragma unroll
  for (int t = 0; t < 8; ++t) oa[t] = v8f{0,0,0,0,0,0,0,0};
  #pragma unroll
  for (int r = 0; r < 8; ++r) { mrow[r] = -3.0e38f; lrow[r] = 0.0f; }

  const int jend = qbase + 15;
  for (int j0 = 0; j0 <= jend; j0 += 32) {
    // ---- batch-load all K fragments, then 8 S WMMAs ----
    v16bf kf[8];
    #pragma unroll
    for (int c = 0; c < 4; ++c) {
      kf[2 * c]     = load_b_frag(kb, j0,      DH, ds0 + c * 32, lane);
      kf[2 * c + 1] = load_b_frag(kb, j0 + 16, DH, ds0 + c * 32, lane);
    }
    v8f s0 = v8f{0,0,0,0,0,0,0,0};
    v8f s1 = v8f{0,0,0,0,0,0,0,0};
    #pragma unroll
    for (int c = 0; c < 4; ++c) {
      s0 = wmma_bf16(qf[c], kf[2 * c],     s0);
      s1 = wmma_bf16(qf[c], kf[2 * c + 1], s1);
    }

    // ---- issue V fragment loads now; consumed after softmax ----
    v16bf vf[8];
    #pragma unroll
    for (int t = 0; t < 8; ++t)
      vf[t] = load_b_frag(vt, ds0 + t * 16, NSEQ, j0, lane);

    // ---- prefetch next kv step into cache (global_prefetch_b8) ----
    if (j0 + 32 <= jend) {
      __builtin_prefetch(kb + (long)(j0 + 32 + lane) * DH + ds0, 0, 1);
      __builtin_prefetch(vt + (long)(ds0 + (lane & 15)) * NSEQ + j0 + 32, 0, 1);
    }

    #pragma unroll
    for (int r = 0; r < 8; ++r) {
      sred[((0 * 8 + r) * 4 + w) * 32 + lane] = s0[r];
      sred[((1 * 8 + r) * 4 + w) * 32 + lane] = s1[r];
    }
    __syncthreads();

    // ---- cross-wave reduction -> full S; scale + causal mask ----
    float sv0[8], sv1[8];
    #pragma unroll
    for (int r = 0; r < 8; ++r) {
      float a0 = 0.f, a1 = 0.f;
      #pragma unroll
      for (int ww = 0; ww < 4; ++ww) {
        a0 += sred[((0 * 8 + r) * 4 + ww) * 32 + lane];
        a1 += sred[((1 * 8 + r) * 4 + ww) * 32 + lane];
      }
      int row = qbase + r + 8 * hi;
      a0 *= scale; a1 *= scale;
      if (j0 + n16 > row)      a0 = MASK_VAL;
      if (j0 + 16 + n16 > row) a1 = MASK_VAL;
      sv0[r] = a0; sv1[r] = a1;
    }

    // ---- online softmax update (reductions across the 16 N-lanes) ----
    #pragma unroll
    for (int r = 0; r < 8; ++r) {
      float rmax = fmaxf(sv0[r], sv1[r]);
      #pragma unroll
      for (int d = 1; d < 16; d <<= 1) rmax = fmaxf(rmax, __shfl_xor(rmax, d, 32));
      float mnew  = fmaxf(mrow[r], rmax);
      float alpha = __expf(mrow[r] - mnew);
      float p0 = __expf(sv0[r] - mnew);
      float p1 = __expf(sv1[r] - mnew);
      float rsum = p0 + p1;
      #pragma unroll
      for (int d = 1; d < 16; d <<= 1) rsum += __shfl_xor(rsum, d, 32);
      lrow[r] = lrow[r] * alpha + rsum;
      mrow[r] = mnew;
      #pragma unroll
      for (int t = 0; t < 8; ++t) oa[t][r] *= alpha;
      if (w == 0) {                         // one wave publishes P (identical in all)
        int m = r + 8 * hi;
        pbuf[m * 32 + n16]      = (__bf16)p0;
        pbuf[m * 32 + 16 + n16] = (__bf16)p1;
      }
    }
    __syncthreads();

    // ---- O_slice += P @ V_slice (A = P from LDS, B = V^T in regs) ----
    {
      int kb8 = hi * 8;
      V16U u;
      u.h[0] = *(const v8bf*)(&pbuf[n16 * 32 + kb8]);
      u.h[1] = *(const v8bf*)(&pbuf[n16 * 32 + 16 + kb8]);
      v16bf pf = u.v;
      #pragma unroll
      for (int t = 0; t < 8; ++t)
        oa[t] = wmma_bf16(pf, vf[t], oa[t]);
    }
    __syncthreads();
  }

  // ---- normalize and store O (bf16, row-major [NSEQ][DH]) ----
  #pragma unroll
  for (int t = 0; t < 8; ++t) {
    #pragma unroll
    for (int r = 0; r < 8; ++r) {
      int row = qbase + r + 8 * hi;
      float val = oa[t][r] / lrow[r];
      ob[(long)row * DH + ds0 + t * 16 + n16] = (__bf16)val;
    }
  }
}

// ---------------- output projection: out = O @ w_out + b_out (f32) ----------------
// Per wave: 32 rows x 64 cols, double-buffered fragment loads.
__global__ void __launch_bounds__(128)
out_proj(const __bf16* __restrict__ ob, const __bf16* __restrict__ wot,
         const float* __restrict__ bias, float* __restrict__ out) {
  const int lane = threadIdx.x & 31;
  const int w    = threadIdx.x >> 5;
  const int rb   = blockIdx.x * 128 + w * 32;
  const int cb   = blockIdx.y * 64;
  const int n16  = lane & 15, hi = lane >> 4;

  v8f acc[2][4];
  #pragma unroll
  for (int mt = 0; mt < 2; ++mt)
    #pragma unroll
    for (int nt = 0; nt < 4; ++nt) acc[mt][nt] = v8f{0,0,0,0,0,0,0,0};

  v16bf a0[2], b0[4], a1[2], b1[4];
  #pragma unroll
  for (int mt = 0; mt < 2; ++mt) a0[mt] = load_a_frag(ob, rb + mt * 16, DH, 0, lane);
  #pragma unroll
  for (int nt = 0; nt < 4; ++nt) b0[nt] = load_b_frag(wot, cb + nt * 16, DH, 0, lane);

  for (int kc = 0; kc < DH; kc += 64) {
    #pragma unroll
    for (int mt = 0; mt < 2; ++mt)
      a1[mt] = load_a_frag(ob, rb + mt * 16, DH, kc + 32, lane);
    #pragma unroll
    for (int nt = 0; nt < 4; ++nt)
      b1[nt] = load_b_frag(wot, cb + nt * 16, DH, kc + 32, lane);
    #pragma unroll
    for (int mt = 0; mt < 2; ++mt)
      #pragma unroll
      for (int nt = 0; nt < 4; ++nt)
        acc[mt][nt] = wmma_bf16(a0[mt], b0[nt], acc[mt][nt]);
    if (kc + 64 < DH) {
      #pragma unroll
      for (int mt = 0; mt < 2; ++mt)
        a0[mt] = load_a_frag(ob, rb + mt * 16, DH, kc + 64, lane);
      #pragma unroll
      for (int nt = 0; nt < 4; ++nt)
        b0[nt] = load_b_frag(wot, cb + nt * 16, DH, kc + 64, lane);
    }
    #pragma unroll
    for (int mt = 0; mt < 2; ++mt)
      #pragma unroll
      for (int nt = 0; nt < 4; ++nt)
        acc[mt][nt] = wmma_bf16(a1[mt], b1[nt], acc[mt][nt]);
  }

  #pragma unroll
  for (int mt = 0; mt < 2; ++mt) {
    #pragma unroll
    for (int nt = 0; nt < 4; ++nt) {
      int col  = cb + nt * 16 + n16;
      float bv = bias[col];
      #pragma unroll
      for (int r = 0; r < 8; ++r) {
        int row = rb + mt * 16 + r + 8 * hi;
        out[(long)row * DIMO + col] = acc[mt][nt][r] + bv;
      }
    }
  }
}

extern "C" void kernel_launch(void* const* d_in, const int* in_sizes, int n_in,
                              void* d_out, int out_size, void* d_ws, size_t ws_size,
                              hipStream_t stream) {
  const float* x     = (const float*)d_in[0];
  const float* w_qkv = (const float*)d_in[1];
  const float* b_qkv = (const float*)d_in[2];
  const float* w_out = (const float*)d_in[3];
  const float* b_out = (const float*)d_in[4];
  float* out = (float*)d_out;

  char* ws = (char*)d_ws;
  __bf16* xb    = (__bf16*)(ws + 0);          // 8192x1024 bf16   = 16 MB
  __bf16* wqkvt = (__bf16*)(ws + 16777216);   // 1536x1024 bf16   =  3 MB (transposed)
  __bf16* woutt = (__bf16*)(ws + 19922944);   // 1024x512  bf16   =  1 MB (transposed)
  __bf16* qb    = (__bf16*)(ws + 20971520);   // 8192x512  bf16   =  8 MB
  __bf16* kbuf  = (__bf16*)(ws + 29360128);   // 8192x512  bf16   =  8 MB
  __bf16* vt    = (__bf16*)(ws + 37748736);   // 512x8192  bf16   =  8 MB (transposed)
  __bf16* obuf  = (__bf16*)(ws + 46137344);   // 8192x512  bf16   =  8 MB

  cvt_bf16_x8<<<(NSEQ * DIMX / 8 + 255) / 256, 256, 0, stream>>>(x, xb, NSEQ * DIMX);
  cvt_transpose_bf16<<<(DIMX * 3 * DH + 255) / 256, 256, 0, stream>>>(w_qkv, wqkvt,
                                                                      DIMX, 3 * DH);
  cvt_transpose_bf16<<<(DH * DIMO + 255) / 256, 256, 0, stream>>>(w_out, woutt,
                                                                  DH, DIMO);

  qkv_proj<<<dim3(NSEQ / 128, (3 * DH) / 64), 128, 0, stream>>>(xb, wqkvt, b_qkv,
                                                                qb, kbuf, vt);
  attn_kernel<<<NSEQ / 16, 128, 0, stream>>>(qb, kbuf, vt, obuf);
  out_proj<<<dim3(NSEQ / 128, DIMO / 64), 128, 0, stream>>>(obuf, woutt, b_out, out);
}
